// GAT_NET_73332271612562
// MI455X (gfx1250) — matrix operand order repaired and checked by reading, not compiled, not run on hardware
//
#include <hip/hip_runtime.h>
#include <math.h>

#define N_NODES 100000
#define N_EDGES 1600000
#define NEG_SLOPE 0.2f

typedef __attribute__((ext_vector_type(16))) __bf16 v16bf;
typedef __attribute__((ext_vector_type(8)))  float v8f;

// ---------- helpers ----------

__device__ __forceinline__ int encodeOrderedF32(float v) {
    int b = __float_as_int(v);
    return (b >= 0) ? b : (b ^ 0x7fffffff);   // monotone map float -> signed int
}
__device__ __forceinline__ float decodeOrderedF32(int s) {
    int b = (s >= 0) ? s : (s ^ 0x7fffffff);
    return __int_as_float(b);
}
__device__ __forceinline__ void atomicMaxF32(int* addr, float v) {
    atomicMax(addr, encodeOrderedF32(v));
}
__device__ __forceinline__ void getEdge(const long long* __restrict__ ei,
                                        long long eid, int& s, int& d) {
    if (eid < (long long)N_EDGES) {
        s = (int)ei[eid];
        d = (int)ei[(long long)N_EDGES + eid];
    } else {
        s = d = (int)(eid - (long long)N_EDGES);   // self loop
    }
}
__device__ __forceinline__ float leaky(float e) {
    return e > 0.f ? e : NEG_SLOPE * e;
}

// ---------- GEMM1: h1[N,64] = x[N,128] @ W1^T (bf16 WMMA, f32 accum) ----------

__global__ __launch_bounds__(256) void gemm1_kernel(const float* __restrict__ x,
                                                    const float* __restrict__ W1,
                                                    float* __restrict__ h1) {
    __shared__ float sx[32 * 128];   // 16 KB node tile
    __shared__ float sw[64 * 128];   // 32 KB full W1
    const int tid = threadIdx.x;
    const int nodeBase = blockIdx.x * 32;

    const size_t xbase = (size_t)nodeBase * 128;
    for (int i = tid; i < 32 * 128; i += 256) sx[i] = x[xbase + i];
    for (int i = tid; i < 64 * 128; i += 256) sw[i] = W1[i];
    __syncthreads();

    const int wave = tid >> 5;
    const int lane = tid & 31;
    const int mt = wave >> 2;        // node sub-tile 0..1
    const int nt = wave & 3;         // col tile 0..3
    const int lm = lane & 15;
    const int hi = lane >> 4;        // lane half

    v8f c = {};
    #pragma unroll
    for (int kc = 0; kc < 4; ++kc) {
        const int k0 = kc * 32;
        v16bf a, b;
        // A 16x32 bf16 layout: lane half 'hi' holds K offsets hi*8 + {0..7, 16..23}
        const float* xrow = &sx[(mt * 16 + lm) * 128 + k0 + hi * 8];
        #pragma unroll
        for (int i = 0; i < 8; ++i)  a[i] = (__bf16)xrow[i];
        #pragma unroll
        for (int i = 8; i < 16; ++i) a[i] = (__bf16)xrow[i + 8];
        // B 32x16 bf16 layout: lane = column n, lane half 'hi' holds K offsets hi*8 + {0..7, 16..23}
        const float* wrow = &sw[(nt * 16 + lm) * 128 + k0 + hi * 8];
        #pragma unroll
        for (int i = 0; i < 8; ++i)  b[i] = (__bf16)wrow[i];
        #pragma unroll
        for (int i = 8; i < 16; ++i) b[i] = (__bf16)wrow[i + 8];
        c = __builtin_amdgcn_wmma_f32_16x16x32_bf16(false, a, false, b,
                                                    (short)0, c, false, false);
    }
    const int col = nt * 16 + lm;
    #pragma unroll
    for (int r = 0; r < 8; ++r) {
        const int m = r + hi * 8;
        h1[(size_t)(nodeBase + mt * 16 + m) * 64 + col] = c[r];
    }
}

// ---------- GEMM2: h2[N,16] = relu(out1)[N,64] @ W2^T (ReLU fused in load) ----------

__global__ __launch_bounds__(256) void gemm2_kernel(const float* __restrict__ out1,
                                                    const float* __restrict__ W2,
                                                    float* __restrict__ h2) {
    __shared__ float sx[128 * 64];   // 32 KB node tile (ReLU applied)
    __shared__ float sw[16 * 64];    // 4 KB full W2
    const int tid = threadIdx.x;
    const int nodeBase = blockIdx.x * 128;

    for (int i = tid; i < 128 * 64; i += 256) {
        const int node = nodeBase + (i >> 6);
        float v = (node < N_NODES) ? out1[(size_t)node * 64 + (i & 63)] : 0.f;
        sx[i] = v > 0.f ? v : 0.f;
    }
    for (int i = tid; i < 16 * 64; i += 256) sw[i] = W2[i];
    __syncthreads();

    const int wave = tid >> 5;
    const int lane = tid & 31;
    const int lm = lane & 15;
    const int hi = lane >> 4;

    v8f c = {};
    #pragma unroll
    for (int kc = 0; kc < 2; ++kc) {
        const int k0 = kc * 32;
        v16bf a, b;
        const float* xrow = &sx[(wave * 16 + lm) * 64 + k0 + hi * 8];
        #pragma unroll
        for (int i = 0; i < 8; ++i)  a[i] = (__bf16)xrow[i];
        #pragma unroll
        for (int i = 8; i < 16; ++i) a[i] = (__bf16)xrow[i + 8];
        const float* wrow = &sw[lm * 64 + k0 + hi * 8];
        #pragma unroll
        for (int i = 0; i < 8; ++i)  b[i] = (__bf16)wrow[i];
        #pragma unroll
        for (int i = 8; i < 16; ++i) b[i] = (__bf16)wrow[i + 8];
        c = __builtin_amdgcn_wmma_f32_16x16x32_bf16(false, a, false, b,
                                                    (short)0, c, false, false);
    }
    #pragma unroll
    for (int r = 0; r < 8; ++r) {
        const int node = nodeBase + wave * 16 + r + hi * 8;
        if (node < N_NODES) h2[(size_t)node * 16 + lm] = c[r];
    }
}

// ---------- per-node attention logits + init (layer 1, H=4) ----------

__global__ void alpha1_kernel(const float* __restrict__ h1,
                              const float* __restrict__ aS, const float* __restrict__ aD,
                              const float* __restrict__ b1,
                              float* __restrict__ as1, float* __restrict__ ad1,
                              int* __restrict__ m1, float* __restrict__ z1,
                              float* __restrict__ out1) {
    const int idx = blockIdx.x * blockDim.x + threadIdx.x;   // node*4 + head
    if (idx >= N_NODES * 4) return;
    const int n = idx >> 2, h = idx & 3;
    const float* hv = &h1[(size_t)n * 64 + h * 16];
    float s = 0.f, d = 0.f;
    #pragma unroll
    for (int f = 0; f < 16; ++f) {
        s = fmaf(hv[f], aS[h * 16 + f], s);
        d = fmaf(hv[f], aD[h * 16 + f], d);
    }
    as1[idx] = s; ad1[idx] = d;
    m1[idx] = (int)0x80000000;   // encoded -inf
    z1[idx] = 0.f;
    #pragma unroll
    for (int f = 0; f < 16; ++f) out1[(size_t)n * 64 + h * 16 + f] = b1[h * 16 + f];
}

// ---------- per-node attention logits + init (layer 2, H=1) ----------

__global__ void alpha2_kernel(const float* __restrict__ h2,
                              const float* __restrict__ aS, const float* __restrict__ aD,
                              const float* __restrict__ b2,
                              float* __restrict__ as2, float* __restrict__ ad2,
                              int* __restrict__ m2, float* __restrict__ z2,
                              float* __restrict__ out) {
    const int n = blockIdx.x * blockDim.x + threadIdx.x;
    if (n >= N_NODES) return;
    const float* hv = &h2[(size_t)n * 16];
    float s = 0.f, d = 0.f;
    #pragma unroll
    for (int f = 0; f < 16; ++f) {
        s = fmaf(hv[f], aS[f], s);
        d = fmaf(hv[f], aD[f], d);
    }
    as2[n] = s; ad2[n] = d;
    m2[n] = (int)0x80000000;
    z2[n] = 0.f;
    #pragma unroll
    for (int f = 0; f < 16; ++f) out[(size_t)n * 16 + f] = b2[f];
}

// ---------- edge passes (generic over H) ----------

__global__ void edge_max_kernel(const long long* __restrict__ ei,
                                const float* __restrict__ as, const float* __restrict__ ad,
                                int* __restrict__ m, int H) {
    const long long total = (long long)(N_EDGES + N_NODES) * H;
    const long long idx = (long long)blockIdx.x * blockDim.x + threadIdx.x;
    if (idx >= total) return;
    const long long eid = idx / H;
    const int h = (int)(idx - eid * H);
    int s, d; getEdge(ei, eid, s, d);
    const float e = leaky(as[(size_t)s * H + h] + ad[(size_t)d * H + h]);
    atomicMaxF32(&m[(size_t)d * H + h], e);
}

__global__ void edge_sum_kernel(const long long* __restrict__ ei,
                                const float* __restrict__ as, const float* __restrict__ ad,
                                const int* __restrict__ m, float* __restrict__ z, int H) {
    const long long total = (long long)(N_EDGES + N_NODES) * H;
    const long long idx = (long long)blockIdx.x * blockDim.x + threadIdx.x;
    if (idx >= total) return;
    const long long eid = idx / H;
    const int h = (int)(idx - eid * H);
    int s, d; getEdge(ei, eid, s, d);
    const float e = leaky(as[(size_t)s * H + h] + ad[(size_t)d * H + h]);
    const float mm = decodeOrderedF32(m[(size_t)d * H + h]);
    atomicAdd(&z[(size_t)d * H + h], expf(e - mm));
}

__global__ void edge_scatter_kernel(const long long* __restrict__ ei,
                                    const float* __restrict__ as, const float* __restrict__ ad,
                                    const int* __restrict__ m, const float* __restrict__ z,
                                    const float* __restrict__ hsrc, float* __restrict__ out,
                                    int H) {
    const long long total = (long long)(N_EDGES + N_NODES) * H;
    const long long idx = (long long)blockIdx.x * blockDim.x + threadIdx.x;
    if (idx >= total) return;
    const long long eid = idx / H;
    const int h = (int)(idx - eid * H);
    int s, d; getEdge(ei, eid, s, d);
    const float e = leaky(as[(size_t)s * H + h] + ad[(size_t)d * H + h]);
    const float mm = decodeOrderedF32(m[(size_t)d * H + h]);
    const float alpha = expf(e - mm) / z[(size_t)d * H + h];
    const float4* hv = (const float4*)&hsrc[((size_t)s * H + h) * 16];
    float* ov = &out[((size_t)d * H + h) * 16];
    #pragma unroll
    for (int q = 0; q < 4; ++q) {
        const float4 v = hv[q];
        atomicAdd(&ov[q * 4 + 0], v.x * alpha);
        atomicAdd(&ov[q * 4 + 1], v.y * alpha);
        atomicAdd(&ov[q * 4 + 2], v.z * alpha);
        atomicAdd(&ov[q * 4 + 3], v.w * alpha);
    }
}

// ---------- final log_softmax over 16 classes (in place) ----------

__global__ void logsoftmax_kernel(float* __restrict__ out) {
    const int n = blockIdx.x * blockDim.x + threadIdx.x;
    if (n >= N_NODES) return;
    float* v = &out[(size_t)n * 16];
    float mx = v[0];
    #pragma unroll
    for (int f = 1; f < 16; ++f) mx = fmaxf(mx, v[f]);
    float sum = 0.f;
    #pragma unroll
    for (int f = 0; f < 16; ++f) sum += expf(v[f] - mx);
    const float lse = mx + logf(sum);
    #pragma unroll
    for (int f = 0; f < 16; ++f) v[f] = v[f] - lse;
}

// ---------- launcher ----------

extern "C" void kernel_launch(void* const* d_in, const int* in_sizes, int n_in,
                              void* d_out, int out_size, void* d_ws, size_t ws_size,
                              hipStream_t stream) {
    const float*     x   = (const float*)d_in[0];
    const long long* ei  = (const long long*)d_in[1];   // int64 in reference
    const float*     W1  = (const float*)d_in[2];
    const float*     aS1 = (const float*)d_in[3];
    const float*     aD1 = (const float*)d_in[4];
    const float*     b1  = (const float*)d_in[5];
    const float*     W2  = (const float*)d_in[6];
    const float*     aS2 = (const float*)d_in[7];
    const float*     aD2 = (const float*)d_in[8];
    const float*     b2  = (const float*)d_in[9];
    float* out = (float*)d_out;

    // workspace carve-up (~66 MB)
    float* ws  = (float*)d_ws;
    float* h1  = ws; ws += (size_t)N_NODES * 64;
    float* o1  = ws; ws += (size_t)N_NODES * 64;
    float* as1 = ws; ws += (size_t)N_NODES * 4;
    float* ad1 = ws; ws += (size_t)N_NODES * 4;
    int*   m1  = (int*)ws; ws += (size_t)N_NODES * 4;
    float* z1  = ws; ws += (size_t)N_NODES * 4;
    float* h2  = ws; ws += (size_t)N_NODES * 16;
    float* as2 = ws; ws += (size_t)N_NODES;
    float* ad2 = ws; ws += (size_t)N_NODES;
    int*   m2  = (int*)ws; ws += (size_t)N_NODES;
    float* z2  = ws; ws += (size_t)N_NODES;

    const long long tot1 = (long long)(N_EDGES + N_NODES) * 4;
    const long long tot2 = (long long)(N_EDGES + N_NODES);
    const unsigned int eb1 = (unsigned int)((tot1 + 255) / 256);
    const unsigned int eb2 = (unsigned int)((tot2 + 255) / 256);

    // ----- layer 1 -----
    gemm1_kernel<<<N_NODES / 32, 256, 0, stream>>>(x, W1, h1);
    alpha1_kernel<<<(N_NODES * 4 + 255) / 256, 256, 0, stream>>>(h1, aS1, aD1, b1,
                                                                 as1, ad1, m1, z1, o1);
    edge_max_kernel<<<eb1, 256, 0, stream>>>(ei, as1, ad1, m1, 4);
    edge_sum_kernel<<<eb1, 256, 0, stream>>>(ei, as1, ad1, m1, z1, 4);
    edge_scatter_kernel<<<eb1, 256, 0, stream>>>(ei, as1, ad1, m1, z1, h1, o1, 4);

    // ----- layer 2 (ReLU fused into gemm2 load) -----
    gemm2_kernel<<<(N_NODES + 127) / 128, 256, 0, stream>>>(o1, W2, h2);
    alpha2_kernel<<<(N_NODES + 255) / 256, 256, 0, stream>>>(h2, aS2, aD2, b2,
                                                             as2, ad2, m2, z2, out);
    edge_max_kernel<<<eb2, 256, 0, stream>>>(ei, as2, ad2, m2, 1);
    edge_sum_kernel<<<eb2, 256, 0, stream>>>(ei, as2, ad2, m2, z2, 1);
    edge_scatter_kernel<<<eb2, 256, 0, stream>>>(ei, as2, ad2, m2, z2, h2, out, 1);

    // ----- log_softmax -----
    logsoftmax_kernel<<<(N_NODES + 255) / 256, 256, 0, stream>>>(out);
}
